// AttentionLayer_24077586661792
// MI455X (gfx1250) — matrix operand order repaired and checked by reading, not compiled, not run on hardware
//
#include <hip/hip_runtime.h>

typedef __attribute__((ext_vector_type(16))) _Float16 v16h;
typedef __attribute__((ext_vector_type(8)))  _Float16 v8h;
typedef __attribute__((ext_vector_type(4)))  _Float16 v4h;
typedef __attribute__((ext_vector_type(8)))  float    v8f;

#define DIM 4096
#define HEAD_DIM 128
#define N_HEADS 32
#define BSZ 2
#define SEQLEN 1024
#define CACHE_LEN 1024
#define TLEN 2048          // CACHE_LEN + SEQLEN
#define MTOK 2048          // BSZ * SEQLEN

// ---------------------------------------------------------------------------
// helpers
// ---------------------------------------------------------------------------
__device__ __forceinline__ v8f vzero8() {
  v8f z;
#pragma unroll
  for (int i = 0; i < 8; ++i) z[i] = 0.0f;
  return z;
}

__device__ __forceinline__ v16h cat8(v8h lo, v8h hi) {
  return __builtin_shufflevector(lo, hi, 0, 1, 2, 3, 4, 5, 6, 7,
                                 8, 9, 10, 11, 12, 13, 14, 15);
}

// A-matrix fragment (16x32 f16), ISA layout:
// lanes 0-15: M=lane,    K chunks [0..7]  and [16..23]
// lanes 16-31: M=lane-16, K chunks [8..15] and [24..31]
__device__ __forceinline__ v16h frag_a(const _Float16* base, int strideH) {
  const int lane = threadIdx.x & 31;
  const int m = lane & 15;
  const int kb = (lane & 16) ? 8 : 0;
  const _Float16* p = base + m * strideH + kb;
  v8h lo = *(const v8h*)p;
  v8h hi = *(const v8h*)(p + 16);
  return cat8(lo, hi);
}

// B-matrix fragment (32x16 f16) from an N-major (transposed) buffer:
// elements 0-7 = B[kb + j][n], 8-15 = B[kb + 16 + (j-8)][n], kb = 0 (lanes 0-15) / 8 (lanes 16-31)
__device__ __forceinline__ v16h frag_b(const _Float16* baseT, int strideH) {
  const int lane = threadIdx.x & 31;
  const int n = lane & 15;
  const int kb = (lane & 16) ? 8 : 0;
  const _Float16* p = baseT + n * strideH + kb;
  v8h lo = *(const v8h*)p;
  v8h hi = *(const v8h*)(p + 16);
  return cat8(lo, hi);
}

__device__ __forceinline__ v8f wmma_f16(v16h a, v16h b, v8f c) {
  return __builtin_amdgcn_wmma_f32_16x16x32_f16(false, a, false, b,
                                                (short)0, c, false, false);
}

// CDNA5 async global->LDS DMA (ASYNCcnt-tracked, no VGPR round trip).
__device__ __forceinline__ void async_ld_b128(unsigned int lds_addr,
                                              unsigned long long gbase,
                                              unsigned int goff) {
  asm volatile("global_load_async_to_lds_b128 %0, %1, %2"
               :: "v"(lds_addr), "v"(goff), "s"(gbase) : "memory");
}
__device__ __forceinline__ void wait_async0() {
  asm volatile("s_wait_asynccnt 0x0" ::: "memory");
}

// reductions within a 16-lane half-wave (C-matrix row lives across 16 lanes)
__device__ __forceinline__ float half_max(float v) {
  v = fmaxf(v, __shfl_xor(v, 1));
  v = fmaxf(v, __shfl_xor(v, 2));
  v = fmaxf(v, __shfl_xor(v, 4));
  v = fmaxf(v, __shfl_xor(v, 8));
  return v;
}
__device__ __forceinline__ float half_sum(float v) {
  v += __shfl_xor(v, 1);
  v += __shfl_xor(v, 2);
  v += __shfl_xor(v, 4);
  v += __shfl_xor(v, 8);
  return v;
}

// ---------------------------------------------------------------------------
// 1) cache copy + f16 conversion (V stored transposed (b,h,d,t))
// ---------------------------------------------------------------------------
__global__ __launch_bounds__(256) void convert_cache_kernel(
    const float* __restrict__ kc, const float* __restrict__ vc,
    float* __restrict__ xk, float* __restrict__ xv,
    _Float16* __restrict__ k16, _Float16* __restrict__ v16t) {
  const size_t i4 = (size_t)blockIdx.x * 256 + threadIdx.x;
  const size_t base = i4 * 4;                 // flat idx into (b,h,1024,128)
  const int d = (int)(base & 127);
  const int t = (int)((base >> 7) & 1023);
  const int bh = (int)(base >> 17);

  const float4 k4 = *(const float4*)(kc + base);
  const float4 v4 = *(const float4*)(vc + base);

  const size_t o = (((size_t)bh * TLEN) + t) * HEAD_DIM + d;
  *(float4*)(xk + o) = k4;
  *(float4*)(xv + o) = v4;

  v4h kh;
  kh[0] = (_Float16)k4.x; kh[1] = (_Float16)k4.y;
  kh[2] = (_Float16)k4.z; kh[3] = (_Float16)k4.w;
  *(v4h*)(k16 + o) = kh;

  const size_t vt = ((size_t)bh * HEAD_DIM + d) * TLEN + t;
  v16t[vt]            = (_Float16)v4.x;
  v16t[vt + TLEN]     = (_Float16)v4.y;
  v16t[vt + 2 * TLEN] = (_Float16)v4.z;
  v16t[vt + 3 * TLEN] = (_Float16)v4.w;
}

// ---------------------------------------------------------------------------
// 2) WMMA GEMM: Y = A @ W^T, 128x128 block tile, K-step 32, 8 waves,
//    LDS double-buffered so global loads of tile k+1 overlap WMMAs of tile k.
//    MODE 0: Q proj (+rope) -> q16 ws
//    MODE 1: K proj (+rope) -> d_out xk (f32) + k16 ws
//    MODE 2: V proj         -> d_out xv (f32) + vT ws (transposed)
//    MODE 3: O proj (A f16, async-DMA staged) -> d_out output (f32)
// ---------------------------------------------------------------------------
template <int MODE>
__global__ __launch_bounds__(256) void gemm_wmma_kernel(
    const float* __restrict__ A32, const _Float16* __restrict__ A16,
    const float* __restrict__ W, const float* __restrict__ rope,
    float* __restrict__ out32, _Float16* __restrict__ o16a,
    _Float16* __restrict__ o16b) {
  constexpr int K = DIM;
  constexpr int N = DIM;
  constexpr int NK = K / 32;
  __shared__ __align__(16) _Float16 As[2][128 * 40];
  __shared__ __align__(16) _Float16 Bs[2][128 * 40];

  const int tid  = threadIdx.x;
  const int lane = tid & 31;
  const int wid  = tid >> 5;
  const int hi8  = (lane & 16) ? 8 : 0;
  const int blockN = blockIdx.x * 128;
  const int blockM = blockIdx.y * 128;
  const int waveM = (wid & 3) * 32;
  const int waveN = (wid >> 2) * 64;

  const unsigned long long a16base = (unsigned long long)(uintptr_t)A16;

  auto load_f32 = [&](const float* src, int rowBase, int k0, float4* f) {
#pragma unroll
    for (int i = 0; i < 4; ++i) {
      const int idx = tid + i * 256;
      const int row = idx >> 3, c4 = (idx & 7) * 4;
      f[i] = *(const float4*)(src + (size_t)(rowBase + row) * K + k0 + c4);
    }
  };
  auto store_cvt = [&](_Float16* dst, const float4* f) {
#pragma unroll
    for (int i = 0; i < 4; ++i) {
      const int idx = tid + i * 256;
      const int row = idx >> 3, c4 = (idx & 7) * 4;
      v4h h;
      h[0] = (_Float16)f[i].x; h[1] = (_Float16)f[i].y;
      h[2] = (_Float16)f[i].z; h[3] = (_Float16)f[i].w;
      *(v4h*)(dst + row * 40 + c4) = h;
    }
  };
  auto issue_asyncA = [&](int k0, int buf) {
#pragma unroll
    for (int i = 0; i < 2; ++i) {
      const int c = tid + i * 256;                // 512 chunks of 8 halfs
      const int row = c >> 2, o8 = (c & 3) * 8;
      async_ld_b128((unsigned int)(uintptr_t)(&As[buf][row * 40 + o8]),
                    a16base,
                    (unsigned int)(((blockM + row) * K + k0 + o8) * 2));
    }
  };

  v8f acc[2][4];
#pragma unroll
  for (int i = 0; i < 2; ++i)
#pragma unroll
    for (int j = 0; j < 4; ++j) acc[i][j] = vzero8();

  // prologue: stage tile 0 into buffer 0
  {
    float4 fb[4];
    load_f32(W, blockN, 0, fb);
    if (MODE != 3) {
      float4 fa[4];
      load_f32(A32, blockM, 0, fa);
      store_cvt(&As[0][0], fa);
    } else {
      issue_asyncA(0, 0);
    }
    store_cvt(&Bs[0][0], fb);
  }

  int cur = 0;
  for (int kt = 0; kt < NK; ++kt) {
    if (MODE == 3) wait_async0();     // tile `cur` async DMA complete (this wave)
    __syncthreads();                  // all waves' tile `cur` stores visible
    const int nxt = cur ^ 1;
    const bool hasn = (kt + 1 < NK);

    // issue next tile's global loads BEFORE the matrix work (latency hiding)
    float4 fa[4], fb[4];
    if (hasn) {
      const int kn = (kt + 1) * 32;
      load_f32(W, blockN, kn, fb);
      if (MODE != 3) load_f32(A32, blockM, kn, fa);
      else issue_asyncA(kn, nxt);
      if (kt + 2 < NK) {              // gfx1250 global_prefetch_b8, tile k+2
        const int prow = tid >> 1;
        __builtin_prefetch(W + (size_t)(blockN + prow) * K + kn + 32, 0, 1);
        if (MODE != 3)
          __builtin_prefetch(A32 + (size_t)(blockM + prow) * K + kn + 32, 0, 1);
      }
    }

    // all fragment loads, then all 8 WMMAs back-to-back
    const v16h a0 = frag_a(&As[cur][(waveM + 0) * 40], 40);
    const v16h a1 = frag_a(&As[cur][(waveM + 16) * 40], 40);
    v16h bf4[4];
#pragma unroll
    for (int j = 0; j < 4; ++j)
      bf4[j] = frag_b(&Bs[cur][(waveN + j * 16) * 40], 40);
#pragma unroll
    for (int j = 0; j < 4; ++j) acc[0][j] = wmma_f16(a0, bf4[j], acc[0][j]);
#pragma unroll
    for (int j = 0; j < 4; ++j) acc[1][j] = wmma_f16(a1, bf4[j], acc[1][j]);

    // drain next tile's registers into the other LDS buffer
    if (hasn) {
      if (MODE != 3) store_cvt(&As[nxt][0], fa);
      store_cvt(&Bs[nxt][0], fb);
    }
    cur = nxt;
  }

  // epilogue
#pragma unroll
  for (int mi = 0; mi < 2; ++mi)
#pragma unroll
    for (int ni = 0; ni < 4; ++ni) {
      const v8f c = acc[mi][ni];
#pragma unroll
      for (int r = 0; r < 8; ++r) {
        float v = c[r];
        const int gm = blockM + waveM + mi * 16 + r + hi8;
        const int gn = blockN + waveN + ni * 16 + (lane & 15);
        if (MODE <= 1) {
          // RoPE: pair partner sits in the adjacent lane (col +/- 1)
          const float prt = __shfl_xor(v, 1);
          const int s = gm & (SEQLEN - 1);
          const int pair = (gn & (HEAD_DIM - 1)) >> 1;
          const float cs = rope[((size_t)s * 64 + pair) * 2 + 0];
          const float sn = rope[((size_t)s * 64 + pair) * 2 + 1];
          v = (gn & 1) ? (prt * sn + v * cs) : (v * cs - prt * sn);
        }
        if (MODE == 3) {
          out32[(size_t)gm * N + gn] = v;
        } else {
          const int s = gm & (SEQLEN - 1), b = gm >> 10;
          const int h = gn >> 7, d = gn & (HEAD_DIM - 1);
          const int bh = b * N_HEADS + h;
          if (MODE == 0) {
            o16a[(((size_t)bh) * SEQLEN + s) * HEAD_DIM + d] = (_Float16)v;
          } else {
            const size_t o = (((size_t)bh) * TLEN + CACHE_LEN + s) * HEAD_DIM + d;
            out32[o] = v;
            if (MODE == 1)
              o16a[o] = (_Float16)v;
            else
              o16b[(((size_t)bh) * HEAD_DIM + d) * TLEN + CACHE_LEN + s] = (_Float16)v;
          }
        }
      }
    }
}

// ---------------------------------------------------------------------------
// 3) causal flash attention. Block = 8 waves sharing one (b,h). K/V tiles are
//    async-DMA'd into double-buffered LDS: the DMA for step t+32 is issued
//    before the WMMAs for step t, hiding the transfer behind matrix work.
// ---------------------------------------------------------------------------
__global__ __launch_bounds__(256) void flash_attn_kernel(
    const _Float16* __restrict__ q16, const _Float16* __restrict__ k16,
    const _Float16* __restrict__ v16t, _Float16* __restrict__ ctx16) {
  __shared__ __align__(16) _Float16 KT[2][32 * 136];   // 32 keys x 128d (pad)
  __shared__ __align__(16) _Float16 VT[2][128 * 40];   // 128d x 32 keys (pad)
  __shared__ __align__(16) _Float16 P[8][16 * 40];     // per-wave P staging

  const int tid  = threadIdx.x;
  const int lane = tid & 31;
  const int wid  = tid >> 5;
  const int hi8  = (lane & 16) ? 8 : 0;
  const int col  = lane & 15;
  const int bh = blockIdx.x;
  const int b  = bh >> 5, h = bh & 31;
  const int blockQ = blockIdx.y * 128;
  const int q0 = blockQ + wid * 16;

  const _Float16* Qp = q16 + (((size_t)bh) * SEQLEN + q0) * HEAD_DIM;
  const unsigned long long Kbase =
      (unsigned long long)(uintptr_t)(k16 + ((size_t)bh) * TLEN * HEAD_DIM);
  const unsigned long long Vbase =
      (unsigned long long)(uintptr_t)(v16t + ((size_t)bh) * HEAD_DIM * TLEN);
  _Float16* Pw = &P[wid][0];

  auto issue_kv = [&](int t0, int buf) {
#pragma unroll
    for (int i = 0; i < 2; ++i) {
      const int c = tid + i * 256;
      const int kr = c >> 4, ko = (c & 15) * 8;  // key row, half offset
      async_ld_b128((unsigned int)(uintptr_t)(&KT[buf][kr * 136 + ko]), Kbase,
                    (unsigned int)(((t0 + kr) * HEAD_DIM + ko) * 2));
      const int vd = c >> 2, vo = (c & 3) * 8;   // d row, key offset
      async_ld_b128((unsigned int)(uintptr_t)(&VT[buf][vd * 40 + vo]), Vbase,
                    (unsigned int)((vd * TLEN + t0 + vo) * 2));
    }
  };

  v16h aq[4];
#pragma unroll
  for (int i = 0; i < 4; ++i) aq[i] = frag_a(Qp + i * 32, HEAD_DIM);

  v8f accd[8];
  float m_s[8], l_s[8];
#pragma unroll
  for (int j = 0; j < 8; ++j) accd[j] = vzero8();
#pragma unroll
  for (int r = 0; r < 8; ++r) { m_s[r] = -3.0e38f; l_s[r] = 0.0f; }

  const float SCALE = 0.08838834764831845f;    // 1/sqrt(128)
  const int qlim = CACHE_LEN + q0;             // + row = absolute q position
  const int t_end = CACHE_LEN + blockQ + 128;  // block-max causal bound

  issue_kv(0, 0);
  int cur = 0;
  for (int t0 = 0; t0 < t_end; t0 += 32) {
    wait_async0();     // this wave's DMA for tile `cur` complete
    __syncthreads();   // all waves done -> tile `cur` in LDS, prev reads done
    const int nxt = cur ^ 1;
    if (t0 + 32 < t_end) issue_kv(t0 + 32, nxt);

    v8f s0 = vzero8(), s1 = vzero8();
#pragma unroll
    for (int kk = 0; kk < 4; ++kk) {
      const v16h b0 = frag_b(&KT[cur][kk * 32], 136);
      const v16h b1 = frag_b(&KT[cur][16 * 136 + kk * 32], 136);
      s0 = wmma_f16(aq[kk], b0, s0);
      s1 = wmma_f16(aq[kk], b1, s1);
    }
    // online softmax per row (row = r + hi8, spread across 16 lanes)
#pragma unroll
    for (int r = 0; r < 8; ++r) {
      const int rowq = qlim + r + hi8;
      const float e0 = (t0 + col      <= rowq) ? s0[r] * SCALE : -3.0e38f;
      const float e1 = (t0 + 16 + col <= rowq) ? s1[r] * SCALE : -3.0e38f;
      const float mx = half_max(fmaxf(e0, e1));
      const float mnew = fmaxf(m_s[r], mx);
      const float alpha = __expf(m_s[r] - mnew);
      const float p0 = __expf(e0 - mnew);
      const float p1 = __expf(e1 - mnew);
      l_s[r] = l_s[r] * alpha + half_sum(p0 + p1);
      m_s[r] = mnew;
#pragma unroll
      for (int j = 0; j < 8; ++j) accd[j][r] = accd[j][r] * alpha;
      const int row = r + hi8;
      Pw[row * 40 + col]      = (_Float16)p0;
      Pw[row * 40 + 16 + col] = (_Float16)p1;
    }
    // P @ V (same-wave LDS RAW: DS ops are in-order per wave)
    const v16h ap = frag_a(Pw, 40);
#pragma unroll
    for (int j = 0; j < 8; ++j) {
      const v16h bv = frag_b(&VT[cur][(j * 16) * 40], 40);
      accd[j] = wmma_f16(ap, bv, accd[j]);
    }
    cur = nxt;
  }

  // write ctx in (b, s, h*128+d) token-major layout for the O projection
#pragma unroll
  for (int j = 0; j < 8; ++j)
#pragma unroll
    for (int r = 0; r < 8; ++r) {
      const int row = r + hi8;
      const float v = accd[j][r] / l_s[r];
      const size_t o = (((size_t)b * SEQLEN) + q0 + row) * DIM +
                       h * HEAD_DIM + j * 16 + col;
      ctx16[o] = (_Float16)v;
    }
}

// ---------------------------------------------------------------------------
extern "C" void kernel_launch(void* const* d_in, const int* in_sizes, int n_in,
                              void* d_out, int out_size, void* d_ws,
                              size_t ws_size, hipStream_t stream) {
  (void)in_sizes; (void)n_in; (void)out_size; (void)ws_size;
  const float* x       = (const float*)d_in[0];
  const float* k_cache = (const float*)d_in[1];
  const float* v_cache = (const float*)d_in[2];
  const float* rope    = (const float*)d_in[3];
  const float* Wq      = (const float*)d_in[4];
  const float* Wk      = (const float*)d_in[5];
  const float* Wv      = (const float*)d_in[6];
  const float* Wo      = (const float*)d_in[7];

  float* out = (float*)d_out;                                 // (2,1024,4096)
  float* xk = out + (size_t)BSZ * SEQLEN * DIM;               // (2,32,2048,128)
  float* xv = xk + (size_t)BSZ * N_HEADS * TLEN * HEAD_DIM;   // (2,32,2048,128)

  char* ws = (char*)d_ws;
  const size_t q16_bytes  = (size_t)BSZ * N_HEADS * SEQLEN * HEAD_DIM * 2;
  const size_t kv16_bytes = (size_t)BSZ * N_HEADS * TLEN * HEAD_DIM * 2;
  _Float16* q16   = (_Float16*)ws;
  _Float16* k16   = (_Float16*)(ws + q16_bytes);
  _Float16* v16t  = (_Float16*)(ws + q16_bytes + kv16_bytes);
  _Float16* ctx16 = (_Float16*)(ws + q16_bytes + 2 * kv16_bytes);

  // 1) cache copy-through + f16 conversion
  convert_cache_kernel<<<8192, 256, 0, stream>>>(k_cache, v_cache, xk, xv,
                                                 k16, v16t);
  // 2-4) Q/K/V projections (+rope for Q,K)
  dim3 gg(DIM / 128, MTOK / 128);
  gemm_wmma_kernel<0><<<gg, 256, 0, stream>>>(x, nullptr, Wq, rope,
                                              nullptr, q16, nullptr);
  gemm_wmma_kernel<1><<<gg, 256, 0, stream>>>(x, nullptr, Wk, rope,
                                              xk, k16, nullptr);
  gemm_wmma_kernel<2><<<gg, 256, 0, stream>>>(x, nullptr, Wv, nullptr,
                                              xv, nullptr, v16t);
  // 5) flash attention
  flash_attn_kernel<<<dim3(BSZ * N_HEADS, SEQLEN / 128), 256, 0, stream>>>(
      q16, k16, v16t, ctx16);
  // 6) output projection
  gemm_wmma_kernel<3><<<gg, 256, 0, stream>>>(nullptr, ctx16, Wo, nullptr,
                                              out, nullptr, nullptr);
}
